// CausalSelfAttention_48773648613995
// MI455X (gfx1250) — matrix operand order repaired and checked by reading, not compiled, not run on hardware
//
#include <hip/hip_runtime.h>
#include <hip/hip_bf16.h>

typedef __attribute__((ext_vector_type(16))) _Float16 v16h;
typedef __attribute__((ext_vector_type(8)))  _Float16 v8h;
typedef __attribute__((ext_vector_type(4)))  _Float16 v4h;
typedef __attribute__((ext_vector_type(8)))  float    v8f;
typedef __attribute__((ext_vector_type(4)))  float    v4f;

constexpr int BATCH = 2;
constexpr int SEQ   = 2048;
constexpr int EMB   = 1024;
constexpr int NH    = 16;
constexpr int HD    = 64;
constexpr int MROWS = BATCH * SEQ; // 4096

// D = A(16x32 f16) x B(32x16 f16) + C(16x16 f32)
static __device__ __forceinline__ v8f wmma16(v8h a0, v8h a1, v8h b0, v8h b1, v8f c) {
  union { v16h v; v8h h[2]; } A, B;
  A.h[0] = a0; A.h[1] = a1;
  B.h[0] = b0; B.h[1] = b1;
  return __builtin_amdgcn_wmma_f32_16x16x32_f16(false, A.v, false, B.v, (short)0, c, false, false);
}

// ---------------------------------------------------------------------------
// Kernel 1: qkv = x @ W_qkv + b_qkv, scattered to Q/K/V f16 [B][H][S][64]
// Block: 128x128 tile of C, 256 threads (8 waves), wave does 64x32, K-step 32.
// ---------------------------------------------------------------------------
__global__ __launch_bounds__(256) void qkv_gemm(
    const float* __restrict__ X, const float* __restrict__ W,
    const float* __restrict__ bias,
    _Float16* __restrict__ Qb, _Float16* __restrict__ Kb, _Float16* __restrict__ Vb) {
  constexpr int N = 3 * EMB;   // 3072
  constexpr int K = EMB;       // 1024
  __shared__ _Float16 At[128 * 32];   // [row][k] f16
  __shared__ _Float16 Bt[128 * 32];   // [n][k]   f16 (transposed)
  const int tid  = threadIdx.x;
  const int lane = tid & 31;
  const int w    = tid >> 5;
  const int wm   = (w & 1) * 64;
  const int wn   = (w >> 1) * 32;
  const int bm   = blockIdx.y * 128;
  const int bn   = blockIdx.x * 128;
  const int lnN  = lane & 15;
  const int half = lane >> 4;

  v8f zero = {};
  v8f acc[4][2];
#pragma unroll
  for (int i = 0; i < 4; ++i)
#pragma unroll
    for (int j = 0; j < 2; ++j) acc[i][j] = zero;

  for (int k0 = 0; k0 < K; k0 += 32) {
    __syncthreads();
    // Prefetch next K-step's tiles into cache (global_prefetch_b8)
    if (k0 + 32 < K) {
      int t = tid & 127;
      if (tid < 128)
        __builtin_prefetch(X + (size_t)(bm + t) * K + (k0 + 32));
      else
        __builtin_prefetch(W + (size_t)(k0 + 32 + (t >> 2)) * N + bn + (t & 3) * 32);
    }
    // A tile: 128x32 f32 -> f16 (1024 float4 chunks, 4 per thread)
#pragma unroll
    for (int c = 0; c < 4; ++c) {
      int ch  = tid + c * 256;
      int row = ch >> 3;
      int kk  = (ch & 7) * 4;
      v4f a = *reinterpret_cast<const v4f*>(X + (size_t)(bm + row) * K + k0 + kk);
      v4h hh;
      hh.x = (_Float16)a.x; hh.y = (_Float16)a.y;
      hh.z = (_Float16)a.z; hh.w = (_Float16)a.w;
      *reinterpret_cast<v4h*>(&At[row * 32 + kk]) = hh;
    }
    // B tile: 32x128 f32 -> Bt[n][k] f16
#pragma unroll
    for (int c = 0; c < 4; ++c) {
      int ch = tid + c * 256;
      int kr = ch >> 5;
      int n0 = (ch & 31) * 4;
      v4f bv = *reinterpret_cast<const v4f*>(W + (size_t)(k0 + kr) * N + bn + n0);
      Bt[(n0 + 0) * 32 + kr] = (_Float16)bv.x;
      Bt[(n0 + 1) * 32 + kr] = (_Float16)bv.y;
      Bt[(n0 + 2) * 32 + kr] = (_Float16)bv.z;
      Bt[(n0 + 3) * 32 + kr] = (_Float16)bv.w;
    }
    __syncthreads();

    v8h a0[4], a1[4], b0[2], b1[2];
#pragma unroll
    for (int fm = 0; fm < 4; ++fm) {
      int row = wm + fm * 16 + lnN;
      a0[fm] = *reinterpret_cast<const v8h*>(&At[row * 32 + 8 * half]);
      a1[fm] = *reinterpret_cast<const v8h*>(&At[row * 32 + 8 * half + 16]);
    }
#pragma unroll
    for (int fn = 0; fn < 2; ++fn) {
      int col = wn + fn * 16 + lnN;
      b0[fn] = *reinterpret_cast<const v8h*>(&Bt[col * 32 + 8 * half]);
      b1[fn] = *reinterpret_cast<const v8h*>(&Bt[col * 32 + 8 * half + 16]);
    }
#pragma unroll
    for (int fm = 0; fm < 4; ++fm)
#pragma unroll
      for (int fn = 0; fn < 2; ++fn)
        acc[fm][fn] = wmma16(a0[fm], a1[fm], b0[fn], b1[fn], acc[fm][fn]);
  }

  // Epilogue: add bias, scatter into Q/K/V f16 head-major buffers
#pragma unroll
  for (int fm = 0; fm < 4; ++fm)
#pragma unroll
    for (int fn = 0; fn < 2; ++fn)
#pragma unroll
      for (int v = 0; v < 8; ++v) {
        int m = bm + wm + fm * 16 + v + 8 * half;
        int n = bn + wn + fn * 16 + lnN;
        float val = acc[fm][fn][v] + bias[n];
        int bb = m >> 11;            // / SEQ
        int s  = m & (SEQ - 1);
        int which = n >> 10;         // 0=q 1=k 2=v
        int d  = n & (EMB - 1);
        int hh = d >> 6;
        int dd = d & (HD - 1);
        _Float16* dst = (which == 0) ? Qb : ((which == 1) ? Kb : Vb);
        dst[((size_t)(bb * NH + hh) * SEQ + s) * HD + dd] = (_Float16)val;
      }
}

// ---------------------------------------------------------------------------
// Kernel 2: flash attention. Block = (b, h, 64 query rows), 4 waves.
// Each wave owns 16 query rows; K/V tiles (32 keys) shared via LDS.
// K tile is staged with CDNA5 async-to-LDS DMA (ASYNCcnt); V goes through
// VGPRs because it is stored transposed for the P@V B-fragments.
// ---------------------------------------------------------------------------
__global__ __launch_bounds__(128) void flash_attn(
    const _Float16* __restrict__ Qb, const _Float16* __restrict__ Kb,
    const _Float16* __restrict__ Vb, _Float16* __restrict__ Ob) {
  __shared__ _Float16 Kt[32 * 64];      // [key][d]
  __shared__ _Float16 Vt[64 * 32];      // [d][key] (transposed)
  __shared__ _Float16 Pt[4][16 * 32];   // per-wave probs tile [q][key]
  const int tid  = threadIdx.x;
  const int lane = tid & 31;
  const int w    = tid >> 5;
  const int lnN  = lane & 15;
  const int half = lane >> 4;
  const int qblk = blockIdx.x;
  const int hh   = blockIdx.y;
  const int b    = blockIdx.z;
  const int q0   = qblk * 64 + w * 16;
  const size_t headoff = (size_t)(b * NH + hh) * SEQ * HD;

  // Per-thread constant addressing for the async K-tile copy (2 chunks of 16B).
  // Low 32 bits of a flat shared pointer are the LDS byte address (aperture rule).
  unsigned kvoff[2], klds[2];
#pragma unroll
  for (int c = 0; c < 2; ++c) {
    int ch  = tid + c * 128;
    int key = ch >> 3;
    int d0  = (ch & 7) * 8;
    kvoff[c] = (unsigned)((key * HD + d0) * sizeof(_Float16));
    klds[c]  = (unsigned)(uintptr_t)&Kt[key * 64 + d0];
  }

  // Q fragments for this wave's 16 rows, pre-scaled by 1/sqrt(hd)=0.125
  v8h qa0[2], qa1[2];
  {
    const _Float16* qrow = Qb + headoff + (size_t)(q0 + lnN) * HD;
#pragma unroll
    for (int f = 0; f < 2; ++f) {
      v8h t0 = *reinterpret_cast<const v8h*>(qrow + f * 32 + 8 * half);
      v8h t1 = *reinterpret_cast<const v8h*>(qrow + f * 32 + 8 * half + 16);
#pragma unroll
      for (int i = 0; i < 8; ++i) {
        t0[i] = t0[i] * (_Float16)0.125f;
        t1[i] = t1[i] * (_Float16)0.125f;
      }
      qa0[f] = t0; qa1[f] = t1;
    }
  }

  v8f zero = {};
  v8f Oacc[4];
  float mrow[8], lrow[8];
#pragma unroll
  for (int i = 0; i < 4; ++i) Oacc[i] = zero;
#pragma unroll
  for (int v = 0; v < 8; ++v) { mrow[v] = -1e30f; lrow[v] = 0.f; }

  const int nkt = 2 * qblk + 2;  // causal: keys 0 .. qblk*64+63
  for (int kt = 0; kt < nkt; ++kt) {
    const int kbase = kt * 32;
    __syncthreads();  // all waves done reading previous K/V tiles

    // K tile: async DMA global->LDS, layout preserved ([key][d])
    {
      const _Float16* kbasep = Kb + headoff + (size_t)kbase * HD;
#pragma unroll
      for (int c = 0; c < 2; ++c)
        asm volatile("global_load_async_to_lds_b128 %0, %1, %2 offset:0"
                     :: "v"(klds[c]), "v"(kvoff[c]), "s"(kbasep) : "memory");
    }
    // V tile: through VGPRs, stored transposed [d][key]
#pragma unroll
    for (int c = 0; c < 2; ++c) {
      int ch  = tid + c * 128;     // 256 chunks of 8 halves
      int key = ch >> 3;
      int d0  = (ch & 7) * 8;
      v8h vv = *reinterpret_cast<const v8h*>(Vb + headoff + (size_t)(kbase + key) * HD + d0);
#pragma unroll
      for (int i = 0; i < 8; ++i) Vt[(d0 + i) * 32 + key] = vv[i];
    }
    asm volatile("s_wait_asynccnt 0" ::: "memory");
    __syncthreads();
    if (kbase > q0 + 15) continue;  // wave fully above causal frontier

    // S = Q @ K^T  (two 16-key chunks, K-dim 64 = 2 wmma each)
    v8f sacc[2];
    sacc[0] = zero; sacc[1] = zero;
#pragma unroll
    for (int nch = 0; nch < 2; ++nch) {
      int keyloc = nch * 16 + lnN;
#pragma unroll
      for (int f = 0; f < 2; ++f) {
        v8h b0 = *reinterpret_cast<const v8h*>(&Kt[keyloc * 64 + f * 32 + 8 * half]);
        v8h b1 = *reinterpret_cast<const v8h*>(&Kt[keyloc * 64 + f * 32 + 8 * half + 16]);
        sacc[nch] = wmma16(qa0[f], qa1[f], b0, b1, sacc[nch]);
      }
    }
    // Causal mask (only needed on/near the diagonal tile)
    if (kbase + 31 > q0) {
#pragma unroll
      for (int nch = 0; nch < 2; ++nch) {
        int kg = kbase + nch * 16 + lnN;
#pragma unroll
        for (int v = 0; v < 8; ++v) {
          int qg = q0 + v + 8 * half;
          if (kg > qg) sacc[nch][v] = -1e30f;
        }
      }
    }
    // Online softmax: per-row (= per vgpr, per half) max & sum via butterflies
    float corr[8];
#pragma unroll
    for (int v = 0; v < 8; ++v) {
      float x = fmaxf(sacc[0][v], sacc[1][v]);
#pragma unroll
      for (int off = 1; off < 16; off <<= 1) x = fmaxf(x, __shfl_xor(x, off, 32));
      float nm = fmaxf(mrow[v], x);
      corr[v] = __expf(mrow[v] - nm);
      mrow[v] = nm;
      float p0 = __expf(sacc[0][v] - nm);
      float p1 = __expf(sacc[1][v] - nm);
      sacc[0][v] = p0; sacc[1][v] = p1;
      float s = p0 + p1;
#pragma unroll
      for (int off = 1; off < 16; off <<= 1) s += __shfl_xor(s, off, 32);
      lrow[v] = lrow[v] * corr[v] + s;
    }
    // Rescale O accumulators, stash P (f16) in this wave's LDS tile
#pragma unroll
    for (int fn = 0; fn < 4; ++fn)
#pragma unroll
      for (int v = 0; v < 8; ++v) Oacc[fn][v] *= corr[v];
#pragma unroll
    for (int nch = 0; nch < 2; ++nch)
#pragma unroll
      for (int v = 0; v < 8; ++v)
        Pt[w][(v + 8 * half) * 32 + nch * 16 + lnN] = (_Float16)sacc[nch][v];
    // wave-local LDS store->load ordering (no block barrier needed)
    asm volatile("s_wait_dscnt 0" ::: "memory");

    // O += P @ V  (A frag = P 16x32, 4 d-chunks of 16)
    v8h pa0 = *reinterpret_cast<const v8h*>(&Pt[w][lnN * 32 + 8 * half]);
    v8h pa1 = *reinterpret_cast<const v8h*>(&Pt[w][lnN * 32 + 8 * half + 16]);
#pragma unroll
    for (int fn = 0; fn < 4; ++fn) {
      int dcol = fn * 16 + lnN;
      v8h b0 = *reinterpret_cast<const v8h*>(&Vt[dcol * 32 + 8 * half]);
      v8h b1 = *reinterpret_cast<const v8h*>(&Vt[dcol * 32 + 8 * half + 16]);
      Oacc[fn] = wmma16(pa0, pa1, b0, b1, Oacc[fn]);
    }
  }

  // Normalize and store attn (f16) in [B*S][EMB] layout, head-interleaved cols
#pragma unroll
  for (int fn = 0; fn < 4; ++fn)
#pragma unroll
    for (int v = 0; v < 8; ++v) {
      int qg = q0 + v + 8 * half;
      float val = Oacc[fn][v] / lrow[v];
      int col = hh * HD + fn * 16 + lnN;
      Ob[(size_t)(b * SEQ + qg) * EMB + col] = (_Float16)val;
    }
}

// ---------------------------------------------------------------------------
// Kernel 3: out = attn @ W_out + b_out  (attn already f16 in workspace)
// ---------------------------------------------------------------------------
__global__ __launch_bounds__(256) void out_gemm(
    const _Float16* __restrict__ A, const float* __restrict__ W,
    const float* __restrict__ bias, float* __restrict__ Out) {
  constexpr int N = EMB;
  constexpr int K = EMB;
  __shared__ _Float16 At[128 * 32];
  __shared__ _Float16 Bt[128 * 32];
  const int tid  = threadIdx.x;
  const int lane = tid & 31;
  const int w    = tid >> 5;
  const int wm   = (w & 1) * 64;
  const int wn   = (w >> 1) * 32;
  const int bm   = blockIdx.y * 128;
  const int bn   = blockIdx.x * 128;
  const int lnN  = lane & 15;
  const int half = lane >> 4;

  v8f zero = {};
  v8f acc[4][2];
#pragma unroll
  for (int i = 0; i < 4; ++i)
#pragma unroll
    for (int j = 0; j < 2; ++j) acc[i][j] = zero;

  for (int k0 = 0; k0 < K; k0 += 32) {
    __syncthreads();
    if (k0 + 32 < K) {
      int t = tid & 127;
      if (tid < 128)
        __builtin_prefetch(A + (size_t)(bm + t) * K + (k0 + 32));
      else
        __builtin_prefetch(W + (size_t)(k0 + 32 + (t >> 2)) * N + bn + (t & 3) * 32);
    }
    // A tile: already f16, 512 v8h chunks, 2 per thread
#pragma unroll
    for (int c = 0; c < 2; ++c) {
      int ch  = tid + c * 256;
      int row = ch >> 2;
      int kk  = (ch & 3) * 8;
      *reinterpret_cast<v8h*>(&At[row * 32 + kk]) =
          *reinterpret_cast<const v8h*>(A + (size_t)(bm + row) * K + k0 + kk);
    }
    // B tile: f32 -> f16, transposed
#pragma unroll
    for (int c = 0; c < 4; ++c) {
      int ch = tid + c * 256;
      int kr = ch >> 5;
      int n0 = (ch & 31) * 4;
      v4f bv = *reinterpret_cast<const v4f*>(W + (size_t)(k0 + kr) * N + bn + n0);
      Bt[(n0 + 0) * 32 + kr] = (_Float16)bv.x;
      Bt[(n0 + 1) * 32 + kr] = (_Float16)bv.y;
      Bt[(n0 + 2) * 32 + kr] = (_Float16)bv.z;
      Bt[(n0 + 3) * 32 + kr] = (_Float16)bv.w;
    }
    __syncthreads();

    v8h a0[4], a1[4], b0[2], b1[2];
#pragma unroll
    for (int fm = 0; fm < 4; ++fm) {
      int row = wm + fm * 16 + lnN;
      a0[fm] = *reinterpret_cast<const v8h*>(&At[row * 32 + 8 * half]);
      a1[fm] = *reinterpret_cast<const v8h*>(&At[row * 32 + 8 * half + 16]);
    }
#pragma unroll
    for (int fn = 0; fn < 2; ++fn) {
      int col = wn + fn * 16 + lnN;
      b0[fn] = *reinterpret_cast<const v8h*>(&Bt[col * 32 + 8 * half]);
      b1[fn] = *reinterpret_cast<const v8h*>(&Bt[col * 32 + 8 * half + 16]);
    }
#pragma unroll
    for (int fm = 0; fm < 4; ++fm)
#pragma unroll
      for (int fn = 0; fn < 2; ++fn)
        acc[fm][fn] = wmma16(a0[fm], a1[fm], b0[fn], b1[fn], acc[fm][fn]);
  }

#pragma unroll
  for (int fm = 0; fm < 4; ++fm)
#pragma unroll
    for (int fn = 0; fn < 2; ++fn)
#pragma unroll
      for (int v = 0; v < 8; ++v) {
        int m = bm + wm + fm * 16 + v + 8 * half;
        int n = bn + wn + fn * 16 + lnN;
        Out[(size_t)m * N + n] = acc[fm][fn][v] + bias[n];
      }
}

// ---------------------------------------------------------------------------
extern "C" void kernel_launch(void* const* d_in, const int* in_sizes, int n_in,
                              void* d_out, int out_size, void* d_ws, size_t ws_size,
                              hipStream_t stream) {
  (void)in_sizes; (void)n_in; (void)out_size; (void)ws_size;
  const float* x    = (const float*)d_in[0];
  const float* Wqkv = (const float*)d_in[1];
  const float* bqkv = (const float*)d_in[2];
  const float* Wout = (const float*)d_in[3];
  const float* bout = (const float*)d_in[4];
  float* out = (float*)d_out;

  // workspace: Q, K, V, attn — each B*S*EMB f16 = 8 MB (32 MB total)
  char* ws = (char*)d_ws;
  const size_t seg = (size_t)BATCH * SEQ * EMB * sizeof(_Float16);
  _Float16* Qb = (_Float16*)(ws);
  _Float16* Kb = (_Float16*)(ws + seg);
  _Float16* Vb = (_Float16*)(ws + 2 * seg);
  _Float16* Ab = (_Float16*)(ws + 3 * seg);

  dim3 g1(3 * EMB / 128, MROWS / 128);        // (24, 32)
  qkv_gemm<<<g1, 256, 0, stream>>>(x, Wqkv, bqkv, Qb, Kb, Vb);

  dim3 g2(SEQ / 64, NH, BATCH);               // (32, 16, 2)
  flash_attn<<<g2, 128, 0, stream>>>(Qb, Kb, Vb, Ab);

  dim3 g3(EMB / 128, MROWS / 128);            // (8, 32)
  out_gemm<<<g3, 256, 0, stream>>>(Ab, Wout, bout, out);
}
